// DeepPMTransformerEncoderLayer_18622978196260
// MI455X (gfx1250) — matrix operand order repaired and checked by reading, not compiled, not run on hardware
//
#include <hip/hip_runtime.h>
#include <hip/hip_bf16.h>
#include <math.h>

// Problem constants (from reference)
#define B_   8
#define L_   1024
#define D_   256
#define H_   8
#define DFF_ 2048
#define DH_  32
#define M_   (B_ * L_)   // 8192 rows

typedef __attribute__((ext_vector_type(16))) _Float16 v16h;
typedef __attribute__((ext_vector_type(8)))  _Float16 v8h;
typedef __attribute__((ext_vector_type(8)))  float    v8f;

// ---------------------------------------------------------------------------
// WMMA helpers (CDNA5: D = A(16x32 f16) x B(32x16 f16) + C(16x16 f32))
// ---------------------------------------------------------------------------
__device__ __forceinline__ v8f wmma_f16(v16h a, v16h b, v8f c) {
  return __builtin_amdgcn_wmma_f32_16x16x32_f16(false, a, false, b, (short)0, c,
                                                false, false);
}

// A-frag: lane (lo,hi) holds row lo, K = hi*16 .. hi*16+15
__device__ __forceinline__ v16h load_afrag(const _Float16* __restrict__ rowp,
                                           int kc, int hi) {
  v8h a0 = *(const v8h*)(rowp + kc + hi * 16);
  v8h a1 = *(const v8h*)(rowp + kc + hi * 16 + 8);
  v16h a;
#pragma unroll
  for (int i = 0; i < 8; ++i) { a[i] = a0[i]; a[8 + i] = a1[i]; }
  return a;
}

// B-frag: lane (lo,hi) holds column lo, K = hi*16 .. hi*16+15 (contiguous in a
// row of W since B[k][n] = W[n][k]).
__device__ __forceinline__ v16h load_bfrag(const _Float16* __restrict__ rowp,
                                           int kc, int hi) {
  return *(const v16h*)(rowp + kc + hi * 16);
}

// Accumulate a 32(M) x 32(N) tile: 2 A-frags x 2 B-frags -> 4 WMMAs per
// K-chunk with only 8 b128 loads (2 loads per wmma).
__device__ __forceinline__ void gemm_acc2(const _Float16* __restrict__ A,
                                          const _Float16* __restrict__ W, int K,
                                          int m0, int n0, int lo, int hi,
                                          v8f& c00, v8f& c01, v8f& c10,
                                          v8f& c11) {
  const _Float16* arow0 = A + (size_t)(m0 + lo) * K;
  const _Float16* arow1 = A + (size_t)(m0 + 16 + lo) * K;
  const _Float16* w0    = W + (size_t)(n0 + lo) * K;
  const _Float16* w1    = W + (size_t)(n0 + 16 + lo) * K;
  for (int kc = 0; kc < K; kc += 32) {
    if (kc + 32 < K) {  // prefetch next K-chunk of shared W rows
      __builtin_prefetch(w0 + kc + 32, 0, 1);
      __builtin_prefetch(w1 + kc + 32, 0, 1);
    }
    v16h a0 = load_afrag(arow0, kc, hi);
    v16h a1 = load_afrag(arow1, kc, hi);
    v16h b0 = load_bfrag(w0, kc, hi);
    v16h b1 = load_bfrag(w1, kc, hi);
    c00 = wmma_f16(a0, b0, c00);
    c01 = wmma_f16(a0, b1, c01);
    c10 = wmma_f16(a1, b0, c10);
    c11 = wmma_f16(a1, b1, c11);
  }
}

// Block = 8 waves as 4(M) x 2(N): block tile 128(M) x 64(N).
#define GEMM_PROLOG()                                                          \
  int tid = threadIdx.x, wid = tid >> 5, lane = tid & 31;                      \
  int lo = lane & 15, hi = lane >> 4;                                          \
  int m0 = blockIdx.y * 128 + (wid & 3) * 32;                                  \
  int n0 = blockIdx.x * 64 + (wid >> 2) * 32;                                  \
  v8f c00 = {}, c01 = {}, c10 = {}, c11 = {};

// ---------------------------------------------------------------------------
// Utility kernels
// ---------------------------------------------------------------------------
__global__ void f32_to_f16_kernel(const float* __restrict__ in,
                                  _Float16* __restrict__ out, int n) {
  int i = blockIdx.x * 256 + threadIdx.x;
  if (i < n) out[i] = (_Float16)in[i];
}

// valid length per batch: first index where pad-mask is true (else L)
__global__ __launch_bounds__(1024)
void seqlen_kernel(const unsigned char* __restrict__ mask,
                   int* __restrict__ slen) {
  __shared__ int red[32];
  int b = blockIdx.x, t = threadIdx.x;
  int v = mask[(size_t)b * L_ + t] ? t : L_;
#pragma unroll
  for (int m = 16; m >= 1; m >>= 1) v = min(v, __shfl_xor(v, m, 32));
  if ((t & 31) == 0) red[t >> 5] = v;
  __syncthreads();
  if (t < 32) {
    int x = red[t];
#pragma unroll
    for (int m = 16; m >= 1; m >>= 1) x = min(x, __shfl_xor(x, m, 32));
    if (t == 0) slen[b] = x;
  }
}

// ---------------------------------------------------------------------------
// GEMM 1: QKV projection. A = srcH [8192,256], W = in_proj_w [768,256].
// Scatter: Q,K -> [B,H,L,DH]; V -> transposed [B,H,DH,L].
// ---------------------------------------------------------------------------
__device__ __forceinline__ void qkv_scatter(float val, int row, int n,
                                            _Float16* q, _Float16* k,
                                            _Float16* vt) {
  int b = row >> 10, l = row & 1023;
  int which = n >> 8, nn = n & 255;
  int hd = nn >> 5, dh = nn & 31;
  size_t bh = (size_t)b * H_ + hd;
  if (which == 0)      q[(bh * L_ + l) * DH_ + dh] = (_Float16)val;
  else if (which == 1) k[(bh * L_ + l) * DH_ + dh] = (_Float16)val;
  else                 vt[(bh * DH_ + dh) * L_ + l] = (_Float16)val;
}

__global__ __launch_bounds__(256)
void gemm_qkv_kernel(const _Float16* __restrict__ A,
                     const _Float16* __restrict__ W,
                     const float* __restrict__ bias,
                     _Float16* __restrict__ q, _Float16* __restrict__ k,
                     _Float16* __restrict__ vt) {
  GEMM_PROLOG();
  gemm_acc2(A, W, D_, m0, n0, lo, hi, c00, c01, c10, c11);
#pragma unroll
  for (int mh = 0; mh < 2; ++mh) {
    v8f& d0 = mh ? c10 : c00;
    v8f& d1 = mh ? c11 : c01;
#pragma unroll
    for (int r = 0; r < 8; ++r) {
      int row = m0 + mh * 16 + r + hi * 8;
      qkv_scatter(d0[r] + bias[n0 + lo],      row, n0 + lo,      q, k, vt);
      qkv_scatter(d1[r] + bias[n0 + 16 + lo], row, n0 + 16 + lo, q, k, vt);
    }
  }
}

// ---------------------------------------------------------------------------
// Flash attention. One wave = 16 q-rows; block = 4 waves = 64 q-rows.
// grid = (B*H, L/64). Streaming softmax over key tiles of 32.
// ---------------------------------------------------------------------------
__global__ __launch_bounds__(128)
void attn_kernel(const _Float16* __restrict__ Q, const _Float16* __restrict__ K,
                 const _Float16* __restrict__ Vt, const int* __restrict__ slen,
                 _Float16* __restrict__ ctx) {
  __shared__ _Float16 psh[4][16 * 32];  // per-wave P staging (C->A relayout)
  int tid = threadIdx.x, wid = tid >> 5, lane = tid & 31;
  int lo = lane & 15, hi = lane >> 4;
  int bh = blockIdx.x;             // b*H + h
  int b = bh >> 3, hd = bh & 7;
  int q0 = blockIdx.y * 64 + wid * 16;
  int s = slen[b];
  _Float16* cbase = ctx + ((size_t)b * L_) * D_ + hd * DH_;

  if (q0 >= s) {  // fully padded tile: emit zeros (final kernel re-zeros too)
#pragma unroll
    for (int r = 0; r < 8; ++r) {
      size_t ro = (size_t)(q0 + r + hi * 8) * D_;
      cbase[ro + lo] = (_Float16)0.f;
      cbase[ro + 16 + lo] = (_Float16)0.f;
    }
    return;
  }

  // Q fragment (rows q0..q0+15, K-dim = DH = 32, exactly one WMMA K-chunk)
  const _Float16* qrow = Q + ((size_t)bh * L_ + q0 + lo) * DH_;
  v16h aq;
  {
    v8h a0 = *(const v8h*)(qrow + hi * 16);
    v8h a1 = *(const v8h*)(qrow + hi * 16 + 8);
#pragma unroll
    for (int i = 0; i < 8; ++i) { aq[i] = a0[i]; aq[8 + i] = a1[i]; }
  }

  const _Float16* kbase = K + (size_t)bh * L_ * DH_;
  const _Float16* vbase = Vt + (size_t)bh * DH_ * L_;
  const float scale = 0.17677669529663687f;  // 1/sqrt(32)
  const float sf = (float)s;

  v8f acc0 = {}, acc1 = {}, mrow, lrow = {};
  float qrf[8];
#pragma unroll
  for (int r = 0; r < 8; ++r) { mrow[r] = -3.0e38f; qrf[r] = (float)(q0 + r + hi * 8); }

  for (int k0 = 0; k0 < s; k0 += 32) {
    // S = Q K^T for 32 keys (two 16-key column tiles)
    v16h bk0 = *(const v16h*)(kbase + (size_t)(k0 + lo) * DH_ + hi * 16);
    v16h bk1 = *(const v16h*)(kbase + (size_t)(k0 + 16 + lo) * DH_ + hi * 16);
    v8f s0 = {}, s1 = {};
    s0 = wmma_f16(aq, bk0, s0);
    s1 = wmma_f16(aq, bk1, s1);

    // scale + distance bias + key padding mask
    float key0 = (float)(k0 + lo), key1 = (float)(k0 + 16 + lo);
    bool ok0 = key0 < sf, ok1 = key1 < sf;
#pragma unroll
    for (int r = 0; r < 8; ++r) {
      float qv = qrf[r];
      s0[r] = ok0 ? (s0[r] * scale + (sf - fabsf(qv - key0)) / sf) : -3.0e38f;
      s1[r] = ok1 ? (s1[r] * scale + (sf - fabsf(qv - key1)) / sf) : -3.0e38f;
    }

    // row max over the 32 keys (16-lane butterfly within each half)
    v8f tmax;
#pragma unroll
    for (int r = 0; r < 8; ++r) tmax[r] = fmaxf(s0[r], s1[r]);
#pragma unroll
    for (int m = 1; m < 16; m <<= 1)
#pragma unroll
      for (int r = 0; r < 8; ++r) tmax[r] = fmaxf(tmax[r], __shfl_xor(tmax[r], m, 32));

    v8f p0, p1, psum;
#pragma unroll
    for (int r = 0; r < 8; ++r) {
      float mn = fmaxf(mrow[r], tmax[r]);
      float fac = __expf(mrow[r] - mn);
      mrow[r] = mn;
      acc0[r] *= fac; acc1[r] *= fac; lrow[r] *= fac;
      p0[r] = __expf(s0[r] - mn);
      p1[r] = __expf(s1[r] - mn);
      psum[r] = p0[r] + p1[r];
    }
#pragma unroll
    for (int m = 1; m < 16; m <<= 1)
#pragma unroll
      for (int r = 0; r < 8; ++r) psum[r] += __shfl_xor(psum[r], m, 32);
#pragma unroll
    for (int r = 0; r < 8; ++r) lrow[r] += psum[r];

    // C-layout P -> LDS -> A-layout P (for P x V)
    _Float16* pl = psh[wid];
#pragma unroll
    for (int r = 0; r < 8; ++r) {
      pl[(r + hi * 8) * 32 + lo]      = (_Float16)p0[r];
      pl[(r + hi * 8) * 32 + 16 + lo] = (_Float16)p1[r];
    }
    asm volatile("s_wait_dscnt 0" ::: "memory");
    v16h ap;
    {
      v8h a0 = *(const v8h*)(pl + lo * 32 + hi * 16);
      v8h a1 = *(const v8h*)(pl + lo * 32 + hi * 16 + 8);
#pragma unroll
      for (int i = 0; i < 8; ++i) { ap[i] = a0[i]; ap[8 + i] = a1[i]; }
    }

    // ctx += P(16x32) x V(32xDH); Vt rows are contiguous over keys
    v16h bv0 = *(const v16h*)(vbase + (size_t)lo * L_ + k0 + hi * 16);
    v16h bv1 = *(const v16h*)(vbase + (size_t)(16 + lo) * L_ + k0 + hi * 16);
    acc0 = wmma_f16(ap, bv0, acc0);
    acc1 = wmma_f16(ap, bv1, acc1);
  }

  // normalize and store ctx (f16, [B,L,D] with head offset)
#pragma unroll
  for (int r = 0; r < 8; ++r) {
    float inv = 1.0f / lrow[r];
    size_t ro = (size_t)(q0 + r + hi * 8) * D_;
    cbase[ro + lo]      = (_Float16)(acc0[r] * inv);
    cbase[ro + 16 + lo] = (_Float16)(acc1[r] * inv);
  }
}

// ---------------------------------------------------------------------------
// Post-attention GEMMs with fused epilogues
// ---------------------------------------------------------------------------
// h_attn = ctx @ out_w^T + out_b  -> f16
__global__ __launch_bounds__(256)
void gemm_out_kernel(const _Float16* __restrict__ A,
                     const _Float16* __restrict__ W,
                     const float* __restrict__ bias,
                     _Float16* __restrict__ out) {
  GEMM_PROLOG();
  gemm_acc2(A, W, D_, m0, n0, lo, hi, c00, c01, c10, c11);
#pragma unroll
  for (int mh = 0; mh < 2; ++mh) {
    v8f& d0 = mh ? c10 : c00;
    v8f& d1 = mh ? c11 : c01;
#pragma unroll
    for (int r = 0; r < 8; ++r) {
      size_t ro = (size_t)(m0 + mh * 16 + r + hi * 8) * D_;
      out[ro + n0 + lo]      = (_Float16)(d0[r] + bias[n0 + lo]);
      out[ro + n0 + 16 + lo] = (_Float16)(d1[r] + bias[n0 + 16 + lo]);
    }
  }
}

// h = src + h_attn @ proj_w^T + proj_b  -> f32 (residual) and f16 (FFN input)
__global__ __launch_bounds__(256)
void gemm_resproj_kernel(const _Float16* __restrict__ A,
                         const _Float16* __restrict__ W,
                         const float* __restrict__ bias,
                         const float* __restrict__ src,
                         float* __restrict__ hF, _Float16* __restrict__ hH) {
  GEMM_PROLOG();
  gemm_acc2(A, W, D_, m0, n0, lo, hi, c00, c01, c10, c11);
#pragma unroll
  for (int mh = 0; mh < 2; ++mh) {
    v8f& d0 = mh ? c10 : c00;
    v8f& d1 = mh ? c11 : c01;
#pragma unroll
    for (int r = 0; r < 8; ++r) {
      size_t ro = (size_t)(m0 + mh * 16 + r + hi * 8) * D_;
      float v0 = src[ro + n0 + lo]      + d0[r] + bias[n0 + lo];
      float v1 = src[ro + n0 + 16 + lo] + d1[r] + bias[n0 + 16 + lo];
      hF[ro + n0 + lo] = v0;       hH[ro + n0 + lo] = (_Float16)v0;
      hF[ro + n0 + 16 + lo] = v1;  hH[ro + n0 + 16 + lo] = (_Float16)v1;
    }
  }
}

// ff = gelu(h @ ff1_w^T + ff1_b)  (exact erf gelu) -> f16 [M, DFF]
__global__ __launch_bounds__(256)
void gemm_ff1_kernel(const _Float16* __restrict__ A,
                     const _Float16* __restrict__ W,
                     const float* __restrict__ bias,
                     _Float16* __restrict__ out) {
  GEMM_PROLOG();
  gemm_acc2(A, W, D_, m0, n0, lo, hi, c00, c01, c10, c11);
#pragma unroll
  for (int mh = 0; mh < 2; ++mh) {
    v8f& d0 = mh ? c10 : c00;
    v8f& d1 = mh ? c11 : c01;
#pragma unroll
    for (int r = 0; r < 8; ++r) {
      size_t ro = (size_t)(m0 + mh * 16 + r + hi * 8) * DFF_;
      float v0 = d0[r] + bias[n0 + lo];
      float v1 = d1[r] + bias[n0 + 16 + lo];
      v0 = 0.5f * v0 * (1.0f + erff(v0 * 0.70710678118654752f));
      v1 = 0.5f * v1 * (1.0f + erff(v1 * 0.70710678118654752f));
      out[ro + n0 + lo]      = (_Float16)v0;
      out[ro + n0 + 16 + lo] = (_Float16)v1;
    }
  }
}

// out = h + ff @ ff2_w^T + ff2_b ; zero padded rows -> f32 d_out
__global__ __launch_bounds__(256)
void gemm_ff2_kernel(const _Float16* __restrict__ A,
                     const _Float16* __restrict__ W,
                     const float* __restrict__ bias,
                     const float* __restrict__ hF,
                     const int* __restrict__ slen, float* __restrict__ out) {
  GEMM_PROLOG();
  gemm_acc2(A, W, DFF_, m0, n0, lo, hi, c00, c01, c10, c11);
#pragma unroll
  for (int mh = 0; mh < 2; ++mh) {
    v8f& d0 = mh ? c10 : c00;
    v8f& d1 = mh ? c11 : c01;
#pragma unroll
    for (int r = 0; r < 8; ++r) {
      int row = m0 + mh * 16 + r + hi * 8;
      int b = row >> 10, l = row & 1023;
      bool pad = l >= slen[b];
      size_t ro = (size_t)row * D_;
      float v0 = hF[ro + n0 + lo]      + d0[r] + bias[n0 + lo];
      float v1 = hF[ro + n0 + 16 + lo] + d1[r] + bias[n0 + 16 + lo];
      out[ro + n0 + lo]      = pad ? 0.0f : v0;
      out[ro + n0 + 16 + lo] = pad ? 0.0f : v1;
    }
  }
}

// ---------------------------------------------------------------------------
// Launch
// ---------------------------------------------------------------------------
extern "C" void kernel_launch(void* const* d_in, const int* in_sizes, int n_in,
                              void* d_out, int out_size, void* d_ws,
                              size_t ws_size, hipStream_t stream) {
  (void)in_sizes; (void)n_in; (void)out_size; (void)ws_size;
  const float* src        = (const float*)d_in[0];
  const unsigned char* km = (const unsigned char*)d_in[1];  // bool mask
  const float* in_proj_w  = (const float*)d_in[2];
  const float* in_proj_b  = (const float*)d_in[3];
  const float* out_w      = (const float*)d_in[4];
  const float* out_b      = (const float*)d_in[5];
  const float* proj_w     = (const float*)d_in[6];
  const float* proj_b     = (const float*)d_in[7];
  const float* ff1_w      = (const float*)d_in[8];
  const float* ff1_b      = (const float*)d_in[9];
  const float* ff2_w      = (const float*)d_in[10];
  const float* ff2_b      = (const float*)d_in[11];
  float* outp = (float*)d_out;

  // workspace carve-up (~74 MB total), 256B aligned
  char* p = (char*)d_ws;
  size_t off = 0;
  auto take = [&](size_t bytes) {
    void* r = p + off;
    off = (off + bytes + 255) & ~(size_t)255;
    return r;
  };
  _Float16* srcH  = (_Float16*)take((size_t)M_ * D_ * 2);
  _Float16* wqkvH = (_Float16*)take((size_t)3 * D_ * D_ * 2);
  _Float16* woutH = (_Float16*)take((size_t)D_ * D_ * 2);
  _Float16* wprjH = (_Float16*)take((size_t)D_ * D_ * 2);
  _Float16* wff1H = (_Float16*)take((size_t)DFF_ * D_ * 2);
  _Float16* wff2H = (_Float16*)take((size_t)D_ * DFF_ * 2);
  _Float16* qH    = (_Float16*)take((size_t)M_ * D_ * 2);
  _Float16* kH    = (_Float16*)take((size_t)M_ * D_ * 2);
  _Float16* vtH   = (_Float16*)take((size_t)M_ * D_ * 2);
  _Float16* ctxH  = (_Float16*)take((size_t)M_ * D_ * 2);
  _Float16* haH   = (_Float16*)take((size_t)M_ * D_ * 2);
  _Float16* hH    = (_Float16*)take((size_t)M_ * D_ * 2);
  _Float16* ffH   = (_Float16*)take((size_t)M_ * DFF_ * 2);
  float*    hF    = (float*)take((size_t)M_ * D_ * 4);
  int*      slen  = (int*)take(B_ * sizeof(int));

  // 0: sequence lengths
  seqlen_kernel<<<B_, 1024, 0, stream>>>(km, slen);

  // 1: f32 -> f16 conversions
  auto cvt = [&](const float* in, _Float16* out, int n) {
    f32_to_f16_kernel<<<(n + 255) / 256, 256, 0, stream>>>(in, out, n);
  };
  cvt(src, srcH, M_ * D_);
  cvt(in_proj_w, wqkvH, 3 * D_ * D_);
  cvt(out_w, woutH, D_ * D_);
  cvt(proj_w, wprjH, D_ * D_);
  cvt(ff1_w, wff1H, DFF_ * D_);
  cvt(ff2_w, wff2H, D_ * DFF_);

  // 2: QKV projection (M=8192, N=768, K=256); block tile 128x64
  gemm_qkv_kernel<<<dim3(12, 64), 256, 0, stream>>>(srcH, wqkvH, in_proj_b,
                                                    qH, kH, vtH);
  // 3: flash attention (grid = B*H x L/64)
  attn_kernel<<<dim3(64, 16), 128, 0, stream>>>(qH, kH, vtH, slen, ctxH);
  // 4: out projection (N=256)
  gemm_out_kernel<<<dim3(4, 64), 256, 0, stream>>>(ctxH, woutH, out_b, haH);
  // 5: residual + proj (N=256)
  gemm_resproj_kernel<<<dim3(4, 64), 256, 0, stream>>>(haH, wprjH, proj_b,
                                                       src, hF, hH);
  // 6: FFN up + gelu (N=2048)
  gemm_ff1_kernel<<<dim3(32, 64), 256, 0, stream>>>(hH, wff1H, ff1_b, ffH);
  // 7: FFN down + residual + pad-zero (N=256, K=2048)
  gemm_ff2_kernel<<<dim3(4, 64), 256, 0, stream>>>(ffH, wff2H, ff2_b, hF,
                                                   slen, outp);
}
